// MiniLLMAttention_50946902065257
// MI455X (gfx1250) — matrix operand order repaired and checked by reading, not compiled, not run on hardware
//
#include <hip/hip_runtime.h>
#include <cstdint>
#include <cstddef>

typedef __bf16 bf16;
typedef __attribute__((ext_vector_type(16))) __bf16 v16bf;
typedef __attribute__((ext_vector_type(8)))  float  v8f;
typedef int i32x4 __attribute__((vector_size(16)));

// ---------------- feature probes ----------------
#if defined(__has_builtin)
#if __has_builtin(__builtin_amdgcn_global_load_async_to_lds_b128) && \
    __has_builtin(__builtin_amdgcn_s_wait_asynccnt)
#define HAVE_ASYNC_LDS 1
#endif
#if __has_builtin(__builtin_amdgcn_sched_group_barrier)
#define HAVE_SGB 1
#endif
#endif
#ifndef HAVE_ASYNC_LDS
#define HAVE_ASYNC_LDS 0
#endif
#ifndef HAVE_SGB
#define HAVE_SGB 0
#endif

// ---------------- problem constants ----------------
constexpr int S_LEN  = 2048;
constexpr int HDIM   = 2048;
constexpr int NHEAD  = 16;
constexpr int NOPE_D = 128;
constexpr int ROPE_D = 64;
constexpr int VD_D   = 128;
constexpr int QD_D   = 192;    // NOPE+ROPE
constexpr int QLR_D  = 1536;
constexpr int KVLR_D = 512;
constexpr int KVA_D  = 576;    // KVLR+ROPE
constexpr int KVB_D  = 4096;   // NH*(NOPE+VD)
constexpr int QBN_D  = 3072;   // NH*QD
constexpr int INDALL = 128;    // INH*IND
constexpr int ITOPK_D= 512;
constexpr float EPSF = 1e-6f;

// ======================================================================
// Generic WMMA GEMM:  C[M,N](f32) = alpha * A[M,K](bf16) * B[N,K]^T(bf16)
// 128x128x32 tiles, 256 threads (8 wave32), wave w -> 16x128 strip.
// Double-buffered LDS; staging via GLOBAL_LOAD_ASYNC_TO_LDS when available.
// OOB tile rows are clamped (not zeroed): a clamped B row only feeds output
// columns that the store mask discards, likewise clamped A rows feed masked
// output rows, so results for in-range C elements are unaffected.
// ======================================================================
#define BM 128
#define BN 128
#define BK 32
#define LSTR 40   // bf16 elements per LDS row; 80 bytes, 16B-aligned rows

__device__ __forceinline__ void stage_tile(const bf16* __restrict__ G, int rclamp,
                                           int ld, int grow0, int kk,
                                           bf16 (*lds)[LSTR], int tid)
{
    #pragma unroll
    for (int i = 0; i < 2; ++i) {
        int c  = tid + i * 256;
        int r  = c >> 2;
        int kc = (c & 3) << 3;               // 8 bf16 = 16 bytes per chunk
        int grow = grow0 + r;
        if (grow > rclamp) grow = rclamp;    // clamp edge rows (see note above)
        const bf16* src = G + (size_t)grow * ld + (kk + kc);
#if HAVE_ASYNC_LDS
        __builtin_amdgcn_global_load_async_to_lds_b128(
            (__attribute__((address_space(1))) i32x4*)(void*)src,
            (__attribute__((address_space(3))) i32x4*)(void*)&lds[r][kc],
            0, 0);
#else
        *reinterpret_cast<uint4*>(&lds[r][kc]) =
            *reinterpret_cast<const uint4*>(src);
#endif
    }
}

__global__ __launch_bounds__(256)
void gemm_nt_kernel(const bf16* __restrict__ A, const bf16* __restrict__ B,
                    float* __restrict__ C, int M, int N, int K,
                    int lda, int ldb, int ldc, float alpha)
{
    __shared__ __attribute__((aligned(16))) bf16 As[2][BM][LSTR];
    __shared__ __attribute__((aligned(16))) bf16 Bs[2][BN][LSTR];

    const int tid  = threadIdx.x;
    const int lane = tid & 31;
    const int wave = tid >> 5;               // 0..7
    const int bm   = blockIdx.x * BM;
    const int bn   = blockIdx.y * BN;

    v8f acc[8];
    #pragma unroll
    for (int i = 0; i < 8; ++i)
        #pragma unroll
        for (int j = 0; j < 8; ++j) acc[i][j] = 0.0f;

    // fragment addressing (k = 16*(lane>>4) + s, same order for A and B)
    const int arow = wave * 16 + (lane & 15);
    const int kb   = (lane < 16) ? 0 : 16;   // A 16x32: lanes 0-15 K0-15
    const int bcol = lane & 15;
    const int kb2  = (lane < 16) ? 0 : 16;   // B 32x16: lanes 0-15 K0-15

    // prologue: stage tile 0 into buffer 0
    stage_tile(A, M - 1, lda, bm, 0, As[0], tid);
    stage_tile(B, N - 1, ldb, bn, 0, Bs[0], tid);
#if HAVE_ASYNC_LDS
    __builtin_amdgcn_s_wait_asynccnt(0);
#endif
    __syncthreads();

    int p = 0;
    for (int kk = 0; kk < K; kk += BK) {
        const bool last = (kk + BK >= K);
        if (!last) {                          // async-stage next tile into 1-p
            stage_tile(A, M - 1, lda, bm, kk + BK, As[p ^ 1], tid);
            stage_tile(B, N - 1, ldb, bn, kk + BK, Bs[p ^ 1], tid);
        }

        // preload A fragment + all 8 B fragments, then burst 8 WMMAs
        v16bf afrag;
        #pragma unroll
        for (int i = 0; i < 8; ++i) {
            afrag[i]     = As[p][arow][kb + i];
            afrag[i + 8] = As[p][arow][kb + 8 + i];
        }
        v16bf bfrag[8];
        #pragma unroll
        for (int nf = 0; nf < 8; ++nf)
            #pragma unroll
            for (int i = 0; i < 16; ++i)
                bfrag[nf][i] = Bs[p][nf * 16 + bcol][kb2 + i];

        #pragma unroll
        for (int nf = 0; nf < 8; ++nf)
            acc[nf] = __builtin_amdgcn_wmma_f32_16x16x32_bf16(
                false, afrag, false, bfrag[nf], (short)0, acc[nf], false, false);

#if HAVE_SGB
        // scheduling directives: batch the 18 LDS fragment reads (2 A-frag +
        // 16 B-frag ds_load_b128) ahead of the 8 WMMAs so the waitcnt pass
        // can cover them with one counted wait and the WMMAs issue
        // back-to-back (see CDNA5 flow/sync 7.5).
        __builtin_amdgcn_sched_group_barrier(0x100, 18, 0); // DS reads
        __builtin_amdgcn_sched_group_barrier(0x008, 8, 0);  // WMMA group
#endif

#if HAVE_ASYNC_LDS
        if (!last) __builtin_amdgcn_s_wait_asynccnt(0);
#endif
        __syncthreads();
        p ^= 1;
    }

    // store D: VGPR r -> M = r + 8*(lane>=16), N = lane&15
    const int crow = bm + wave * 16 + ((lane < 16) ? 0 : 8);
    #pragma unroll
    for (int nf = 0; nf < 8; ++nf) {
        int col = bn + nf * 16 + bcol;
        if (col < N) {
            #pragma unroll
            for (int r = 0; r < 8; ++r) {
                int row = crow + r;
                if (row < M) C[(size_t)row * ldc + col] = alpha * acc[nf][r];
            }
        }
    }
}

// ======================================================================
// Elementwise fp32 -> bf16
// ======================================================================
__global__ __launch_bounds__(256)
void f32_to_bf16_kernel(const float* __restrict__ in, bf16* __restrict__ out, size_t n)
{
    size_t i = (size_t)blockIdx.x * 256u + threadIdx.x;
    size_t stride = (size_t)gridDim.x * 256u;
    for (; i < n; i += stride) out[i] = (bf16)in[i];
}

// ======================================================================
// RMSNorm: per row, fp32 in (row stride istride, C cols) -> bf16 out
// ======================================================================
__global__ __launch_bounds__(256)
void rmsnorm_kernel(const float* __restrict__ in, int istride, int C,
                    const float* __restrict__ w, bf16* __restrict__ out, int ostride)
{
    __shared__ float red[8];
    int row = blockIdx.x, tid = threadIdx.x;
    const float* x = in + (size_t)row * istride;
    float ss = 0.0f;
    for (int c = tid; c < C; c += 256) { float v = x[c]; ss += v * v; }
    #pragma unroll
    for (int o = 16; o > 0; o >>= 1) ss += __shfl_down(ss, o);
    if ((tid & 31) == 0) red[tid >> 5] = ss;
    __syncthreads();
    if (tid < 32) {
        float v = (tid < 8) ? red[tid] : 0.0f;
        #pragma unroll
        for (int o = 4; o > 0; o >>= 1) v += __shfl_down(v, o);
        if (tid == 0) red[0] = v;
    }
    __syncthreads();
    float inv = rsqrtf(red[0] / (float)C + EPSF);
    bf16* o = out + (size_t)row * ostride;
    for (int c = tid; c < C; c += 256) o[c] = (bf16)(x[c] * inv * w[c]);
}

// ======================================================================
// RoPE + per-head packing
// ======================================================================
__global__ __launch_bounds__(QD_D)
void rope_pack_q_kernel(const float* __restrict__ q, const float* __restrict__ cs,
                        const float* __restrict__ sn, bf16* __restrict__ qf)
{
    int s = blockIdx.x, h = blockIdx.y, d = threadIdx.x;
    const float* qr = q + (size_t)s * QBN_D + h * QD_D;
    float v;
    if (d < NOPE_D) {
        v = qr[d];
    } else {
        int r = d - NOPE_D;
        float x   = qr[d];
        float rot = (r < ROPE_D / 2) ? -qr[NOPE_D + r + ROPE_D / 2]
                                     :  qr[NOPE_D + r - ROPE_D / 2];
        v = x * cs[s * ROPE_D + r] + rot * sn[s * ROPE_D + r];
    }
    qf[((size_t)h * S_LEN + s) * QD_D + d] = (bf16)v;
}

__global__ __launch_bounds__(QD_D)
void rope_pack_k_kernel(const float* __restrict__ kv, const float* __restrict__ kva,
                        const float* __restrict__ cs, const float* __restrict__ sn,
                        bf16* __restrict__ kf)
{
    int s = blockIdx.x, h = blockIdx.y, d = threadIdx.x;
    float v;
    if (d < NOPE_D) {
        v = kv[(size_t)s * KVB_D + h * (NOPE_D + VD_D) + d];
    } else {
        int r = d - NOPE_D;
        const float* kp = kva + (size_t)s * KVA_D + KVLR_D;
        float x   = kp[r];
        float rot = (r < ROPE_D / 2) ? -kp[r + ROPE_D / 2] : kp[r - ROPE_D / 2];
        v = x * cs[s * ROPE_D + r] + rot * sn[s * ROPE_D + r];
    }
    kf[((size_t)h * S_LEN + s) * QD_D + d] = (bf16)v;
}

// vt[h][d][s] = kv[s][h*256 + 128 + d]   (transposed V, so attn@V is NT GEMM)
__global__ __launch_bounds__(VD_D)
void pack_vt_kernel(const float* __restrict__ kv, bf16* __restrict__ vt)
{
    int s = blockIdx.x, h = blockIdx.y, d = threadIdx.x;
    vt[((size_t)h * VD_D + d) * S_LEN + s] =
        (bf16)kv[(size_t)s * KVB_D + h * (NOPE_D + VD_D) + NOPE_D + d];
}

// ======================================================================
// Top-k (512) radix select per row of iscores -> bitmask [S][64] words
// ======================================================================
__device__ __forceinline__ unsigned fkey(float x) {
    unsigned u = __float_as_uint(x);
    return (u & 0x80000000u) ? ~u : (u | 0x80000000u);   // monotonic, bigger=larger
}

__global__ __launch_bounds__(256)
void topk_mask_kernel(const float* __restrict__ isc, unsigned* __restrict__ maskbits)
{
    __shared__ unsigned hist[256];
    __shared__ unsigned sh_prefix;
    __shared__ int      sh_remaining;
    __shared__ int      tiecnt;
    __shared__ unsigned bits[S_LEN / 32];

    int row = blockIdx.x, tid = threadIdx.x;
    const float* src = isc + (size_t)row * S_LEN;
    if (tid == 0) { sh_prefix = 0u; sh_remaining = ITOPK_D; }

    for (int pass = 3; pass >= 0; --pass) {
        hist[tid] = 0u;
        __syncthreads();
        unsigned pref  = sh_prefix;
        int      shift = pass * 8;
        unsigned pmask = (pass == 3) ? 0u : (0xFFFFFFFFu << (shift + 8));
        for (int k = tid; k < S_LEN; k += 256) {
            unsigned u = fkey(src[k]);
            if ((u & pmask) == pref)
                atomicAdd(&hist[(u >> shift) & 255u], 1u);
        }
        __syncthreads();
        if (tid == 0) {
            int rem = sh_remaining;
            unsigned digit = 0;
            for (int d = 255; d >= 0; --d) {
                int c = (int)hist[d];
                if (rem <= c) { digit = (unsigned)d; break; }
                rem -= c;
                if (d == 0) digit = 0;
            }
            sh_prefix    = pref | (digit << shift);
            sh_remaining = rem;
        }
        __syncthreads();
    }

    if (tid == 0) tiecnt = sh_remaining;
    for (int i = tid; i < S_LEN / 32; i += 256) bits[i] = 0u;
    __syncthreads();

    unsigned T = sh_prefix;
    for (int k = tid; k < S_LEN; k += 256) {
        unsigned u = fkey(src[k]);
        bool take = false;
        if (u > T)       take = true;
        else if (u == T) take = (atomicSub(&tiecnt, 1) > 0);
        if (take) atomicOr(&bits[k >> 5], 1u << (k & 31));
    }
    __syncthreads();
    for (int i = tid; i < S_LEN / 32; i += 256)
        maskbits[(size_t)row * (S_LEN / 32) + i] = bits[i];
}

// ======================================================================
// Masked softmax (causal AND topk-allowed), fp32 scores -> bf16 attn
// ======================================================================
__global__ __launch_bounds__(256)
void softmax_mask_kernel(const float* __restrict__ sc, const unsigned* __restrict__ maskbits,
                         bf16* __restrict__ attn)
{
    __shared__ float red[8];
    int q = blockIdx.x, tid = threadIdx.x;
    const float* srow = sc + (size_t)q * S_LEN;
    const unsigned* mb = maskbits + (size_t)q * (S_LEN / 32);

    float mx = -3.0e38f;
    for (int k = tid; k < S_LEN; k += 256) {
        bool ok = (k <= q) && ((mb[k >> 5] >> (k & 31)) & 1u);
        float v = ok ? srow[k] : -1.0e9f;
        mx = fmaxf(mx, v);
    }
    #pragma unroll
    for (int o = 16; o > 0; o >>= 1) mx = fmaxf(mx, __shfl_down(mx, o));
    if ((tid & 31) == 0) red[tid >> 5] = mx;
    __syncthreads();
    if (tid < 32) {
        float v = (tid < 8) ? red[tid] : -3.0e38f;
        #pragma unroll
        for (int o = 4; o > 0; o >>= 1) v = fmaxf(v, __shfl_down(v, o));
        if (tid == 0) red[0] = v;
    }
    __syncthreads();
    mx = red[0];
    __syncthreads();

    float sum = 0.0f;
    for (int k = tid; k < S_LEN; k += 256) {
        bool ok = (k <= q) && ((mb[k >> 5] >> (k & 31)) & 1u);
        float v = ok ? srow[k] : -1.0e9f;
        sum += __expf(v - mx);
    }
    #pragma unroll
    for (int o = 16; o > 0; o >>= 1) sum += __shfl_down(sum, o);
    if ((tid & 31) == 0) red[tid >> 5] = sum;
    __syncthreads();
    if (tid < 32) {
        float v = (tid < 8) ? red[tid] : 0.0f;
        #pragma unroll
        for (int o = 4; o > 0; o >>= 1) v += __shfl_down(v, o);
        if (tid == 0) red[0] = v;
    }
    __syncthreads();
    float inv = 1.0f / red[0];

    for (int k = tid; k < S_LEN; k += 256) {
        bool ok = (k <= q) && ((mb[k >> 5] >> (k & 31)) & 1u);
        float v = ok ? srow[k] : -1.0e9f;
        attn[(size_t)q * S_LEN + k] = (bf16)(__expf(v - mx) * inv);
    }
}

// ======================================================================
// host-side orchestration
// ======================================================================
static inline void launch_gemm(hipStream_t st, const bf16* A, const bf16* B, float* C,
                               int M, int N, int K, int lda, int ldb, int ldc, float alpha)
{
    dim3 g((M + BM - 1) / BM, (N + BN - 1) / BN);
    gemm_nt_kernel<<<g, 256, 0, st>>>(A, B, C, M, N, K, lda, ldb, ldc, alpha);
}

static inline void launch_cvt(hipStream_t st, const float* in, bf16* out, size_t n)
{
    unsigned blocks = (unsigned)((n + 255) / 256);
    if (blocks > 4096u) blocks = 4096u;
    f32_to_bf16_kernel<<<blocks, 256, 0, st>>>(in, out, n);
}

extern "C" void kernel_launch(void* const* d_in, const int* in_sizes, int n_in,
                              void* d_out, int out_size, void* d_ws, size_t ws_size,
                              hipStream_t stream)
{
    const float* x        = (const float*)d_in[0];
    const float* cosb     = (const float*)d_in[1];
    const float* sinb     = (const float*)d_in[2];
    const float* q_a_w    = (const float*)d_in[3];
    const float* q_a_nw   = (const float*)d_in[4];
    const float* q_b_w    = (const float*)d_in[5];
    const float* kv_a_w   = (const float*)d_in[6];
    const float* kv_a_nw  = (const float*)d_in[7];
    const float* kv_b_w   = (const float*)d_in[8];
    const float* o_w      = (const float*)d_in[9];
    const float* idx_q_w  = (const float*)d_in[10];
    const float* idx_k_w  = (const float*)d_in[11];
    float* outp = (float*)d_out;

    // ---- workspace carve-up (needs ~220 MB) ----
    char*  base = (char*)d_ws;
    size_t off  = 0;
    auto alloc = [&](size_t bytes) -> void* {
        void* p = base + off;
        off = (off + bytes + 255) & ~(size_t)255;
        return p;
    };
    bf16* xb    = (bf16*)alloc((size_t)S_LEN * HDIM * 2);
    bf16* wqa   = (bf16*)alloc((size_t)QLR_D * HDIM * 2);
    bf16* wqb   = (bf16*)alloc((size_t)QBN_D * QLR_D * 2);
    bf16* wkva  = (bf16*)alloc((size_t)KVA_D * HDIM * 2);
    bf16* wkvb  = (bf16*)alloc((size_t)KVB_D * KVLR_D * 2);
    bf16* wo    = (bf16*)alloc((size_t)HDIM * HDIM * 2);
    bf16* wiq   = (bf16*)alloc((size_t)INDALL * HDIM * 2);
    bf16* wik   = (bf16*)alloc((size_t)INDALL * HDIM * 2);
    float* qa_f = (float*)alloc((size_t)S_LEN * QLR_D * 4);
    bf16*  qa_n = (bf16*)alloc((size_t)S_LEN * QLR_D * 2);
    float* q_f  = (float*)alloc((size_t)S_LEN * QBN_D * 4);
    float* kva_f= (float*)alloc((size_t)S_LEN * KVA_D * 4);
    bf16*  ckv_n= (bf16*)alloc((size_t)S_LEN * KVLR_D * 2);
    float* kv_f = (float*)alloc((size_t)S_LEN * KVB_D * 4);
    bf16*  qf_b = (bf16*)alloc((size_t)NHEAD * S_LEN * QD_D * 2);
    bf16*  kf_b = (bf16*)alloc((size_t)NHEAD * S_LEN * QD_D * 2);
    bf16*  vt_b = (bf16*)alloc((size_t)NHEAD * VD_D * S_LEN * 2);
    float* iq_f = (float*)alloc((size_t)S_LEN * INDALL * 4);
    float* ik_f = (float*)alloc((size_t)S_LEN * INDALL * 4);
    bf16*  iqb  = (bf16*)alloc((size_t)S_LEN * INDALL * 2);
    bf16*  ikb  = (bf16*)alloc((size_t)S_LEN * INDALL * 2);
    float* isc  = (float*)alloc((size_t)S_LEN * S_LEN * 4);
    unsigned* mbits = (unsigned*)alloc((size_t)S_LEN * (S_LEN / 32) * 4);
    float* sc   = (float*)alloc((size_t)S_LEN * S_LEN * 4);   // per-head, reused
    bf16*  attnb= (bf16*)alloc((size_t)S_LEN * S_LEN * 2);    // per-head, reused
    float* ao_f = (float*)alloc((size_t)S_LEN * NHEAD * VD_D * 4);
    bf16*  ao_b = (bf16*)alloc((size_t)S_LEN * NHEAD * VD_D * 2);
    (void)ws_size; (void)n_in; (void)in_sizes; (void)out_size;

    const float SCALE  = 0.07216878364870323f;   // 192^-0.5
    const float IALPHA = 0.04419417382415922f;   // (32^-0.5) / 4  (mean over 4 idx heads)

    // ---- fp32 -> bf16 staging ----
    launch_cvt(stream, x,       xb,   (size_t)S_LEN * HDIM);
    launch_cvt(stream, q_a_w,   wqa,  (size_t)QLR_D * HDIM);
    launch_cvt(stream, q_b_w,   wqb,  (size_t)QBN_D * QLR_D);
    launch_cvt(stream, kv_a_w,  wkva, (size_t)KVA_D * HDIM);
    launch_cvt(stream, kv_b_w,  wkvb, (size_t)KVB_D * KVLR_D);
    launch_cvt(stream, o_w,     wo,   (size_t)HDIM * HDIM);
    launch_cvt(stream, idx_q_w, wiq,  (size_t)INDALL * HDIM);
    launch_cvt(stream, idx_k_w, wik,  (size_t)INDALL * HDIM);

    // ---- Q path: q_a -> rmsnorm -> q_b ----
    launch_gemm(stream, xb,   wqa, qa_f, S_LEN, QLR_D, HDIM,  HDIM,  HDIM,  QLR_D, 1.0f);
    rmsnorm_kernel<<<S_LEN, 256, 0, stream>>>(qa_f, QLR_D, QLR_D, q_a_nw, qa_n, QLR_D);
    launch_gemm(stream, qa_n, wqb, q_f,  S_LEN, QBN_D, QLR_D, QLR_D, QLR_D, QBN_D, 1.0f);

    // ---- KV path: kv_a -> rmsnorm(first 512) -> kv_b ----
    launch_gemm(stream, xb, wkva, kva_f, S_LEN, KVA_D, HDIM, HDIM, HDIM, KVA_D, 1.0f);
    rmsnorm_kernel<<<S_LEN, 256, 0, stream>>>(kva_f, KVA_D, KVLR_D, kv_a_nw, ckv_n, KVLR_D);
    launch_gemm(stream, ckv_n, wkvb, kv_f, S_LEN, KVB_D, KVLR_D, KVLR_D, KVLR_D, KVB_D, 1.0f);

    // ---- RoPE + per-head packing ----
    {
        dim3 g(S_LEN, NHEAD);
        rope_pack_q_kernel<<<g, QD_D, 0, stream>>>(q_f, cosb, sinb, qf_b);
        rope_pack_k_kernel<<<g, QD_D, 0, stream>>>(kv_f, kva_f, cosb, sinb, kf_b);
        pack_vt_kernel<<<g, VD_D, 0, stream>>>(kv_f, vt_b);
    }

    // ---- indexer: iscores = (iq @ ik^T) * ISCALE/INH ; top-512 mask ----
    launch_gemm(stream, xb, wiq, iq_f, S_LEN, INDALL, HDIM, HDIM, HDIM, INDALL, 1.0f);
    launch_gemm(stream, xb, wik, ik_f, S_LEN, INDALL, HDIM, HDIM, HDIM, INDALL, 1.0f);
    launch_cvt(stream, iq_f, iqb, (size_t)S_LEN * INDALL);
    launch_cvt(stream, ik_f, ikb, (size_t)S_LEN * INDALL);
    launch_gemm(stream, iqb, ikb, isc, S_LEN, S_LEN, INDALL, INDALL, INDALL, S_LEN, IALPHA);
    topk_mask_kernel<<<S_LEN, 256, 0, stream>>>(isc, mbits);

    // ---- per-head attention: scores -> masked softmax -> attn @ V ----
    for (int h = 0; h < NHEAD; ++h) {
        const bf16* qh = qf_b + (size_t)h * S_LEN * QD_D;
        const bf16* kh = kf_b + (size_t)h * S_LEN * QD_D;
        const bf16* vh = vt_b + (size_t)h * VD_D * S_LEN;
        launch_gemm(stream, qh, kh, sc, S_LEN, S_LEN, QD_D, QD_D, QD_D, S_LEN, SCALE);
        softmax_mask_kernel<<<S_LEN, 256, 0, stream>>>(sc, mbits, attnb);
        launch_gemm(stream, attnb, vh, ao_f + (size_t)h * VD_D,
                    S_LEN, VD_D, S_LEN, S_LEN, S_LEN, NHEAD * VD_D, 1.0f);
    }

    // ---- output projection ----
    launch_cvt(stream, ao_f, ao_b, (size_t)S_LEN * NHEAD * VD_D);
    launch_gemm(stream, ao_b, wo, outp, S_LEN, HDIM, NHEAD * VD_D,
                NHEAD * VD_D, NHEAD * VD_D, HDIM, 1.0f);
}
